// Inter_EPI_Transformer_12146167513520
// MI455X (gfx1250) — matrix operand order repaired and checked by reading, not compiled, not run on hardware
//
#include <hip/hip_runtime.h>

#define DEVI __device__ __forceinline__

typedef __attribute__((ext_vector_type(16))) __bf16 v16bf;
typedef __attribute__((ext_vector_type(8)))  float  v8f;

namespace {

constexpr int kB = 2, kC = 4, kN = 224, kV = 7, kW = 256;
constexpr int kS = kV * kW;          // 1792 spatial tokens
constexpr int kEMB = 256, kH = 4, kD = 64;
constexpr int kTD = kC * kN;         // 896 token dim
constexpr int kHF = 10;
constexpr int kRows = kB * kS;       // 3584

DEVI __bf16 f2bf(float f) {
  unsigned u = __builtin_bit_cast(unsigned, f);
  u += 0x7fffu + ((u >> 16) & 1u);                 // round-to-nearest-even
  unsigned short s = (unsigned short)(u >> 16);
  return __builtin_bit_cast(__bf16, s);
}

DEVI float gelu_exact(float x) { return 0.5f * x * (1.f + erff(x * 0.7071067811865475f)); }

// A fragment: 16x32 bf16 from row-major A (ld=lda). ISA 7.12.2 layout:
// lane holds row (lane&15); K = (lane>=16?8:0)+e for e<8, +16 for e>=8.
DEVI v16bf load_a_bf(const __bf16* A, int lda, int m0, int k0, int lane) {
  const __bf16* p = A + (size_t)(m0 + (lane & 15)) * lda + k0 + ((lane >> 4) & 1) * 8;
  v16bf a;
#pragma unroll
  for (int i = 0; i < 8; ++i) { a[i] = p[i]; a[8 + i] = p[16 + i]; }
  return a;
}

DEVI v16bf load_a_f32(const float* A, size_t lda, int m0, int k0, int lane) {
  const float* p = A + (size_t)(m0 + (lane & 15)) * lda + k0 + ((lane >> 4) & 1) * 8;
  v16bf a;
#pragma unroll
  for (int i = 0; i < 8; ++i) { a[i] = f2bf(p[i]); a[8 + i] = f2bf(p[16 + i]); }
  return a;
}

// B fragment: 32x16 bf16, source is B^T stored row-major [n][k] (ld=ldb).
// lane holds column (lane&15); K = (lane>=16?8:0)+e for e<8, +16 for e>=8.
DEVI v16bf load_bt_bf(const __bf16* Bt, int ldb, int n0, int k0, int lane) {
  const __bf16* p = Bt + (size_t)(n0 + (lane & 15)) * ldb + k0 + ((lane >> 4) & 1) * 8;
  v16bf b;
#pragma unroll
  for (int i = 0; i < 8; ++i) { b[i] = p[i]; b[8 + i] = p[16 + i]; }
  return b;
}

DEVI v8f wmma_bf16(v16bf a, v16bf b, v8f c) {
  return __builtin_amdgcn_wmma_f32_16x16x32_bf16(false, a, false, b, (short)0, c, false, false);
}

// ---------- weight / layout prep ----------

__global__ void k_cvt3(const float* a, int na, const float* b, int nb, const float* c, int nc,
                       __bf16* oa, __bf16* ob, __bf16* oc) {
  int i = blockIdx.x * blockDim.x + threadIdx.x;
  if (i < na) oa[i] = f2bf(a[i]);
  if (i < nb) ob[i] = f2bf(b[i]);
  if (i < nc) oc[i] = f2bf(c[i]);
}

// buffer [B,896,S] -> tokT bf16 [B*S, 896]  (token vectors contiguous)
__global__ void k_tokT(const float* buf, __bf16* tokT) {
  int i = blockIdx.x * blockDim.x + threadIdx.x;
  if (i >= kRows * kTD) return;
  int kk = i % kTD, row = i / kTD;
  int b = row / kS, s = row % kS;
  tokT[i] = f2bf(buf[((size_t)b * kTD + kk) * kS + s]);
}

// ---------- GEMM kernels (one wave per 16x16 / 16x64 tile) ----------

// X[b*S+s, e] = tokT[row,:] . w_in[e,:]   (M=3584, N=256, K=896)
__global__ void k_gemm_lin_in(const __bf16* A, const __bf16* Bt, float* X) {
  int lane = threadIdx.x;
  int m0 = blockIdx.x * 16, n0 = blockIdx.y * 16;
  v8f acc = {};
  for (int kc = 0; kc < kTD; kc += 32)
    acc = wmma_bf16(load_a_bf(A, kTD, m0, kc, lane), load_bt_bf(Bt, kTD, n0, kc, lane), acc);
  int col = lane & 15, rb = ((lane >> 4) & 1) * 8;
#pragma unroll
  for (int r = 0; r < 8; ++r) X[(size_t)(m0 + rb + r) * kEMB + n0 + col] = acc[r];
}

// LayerNorm over EMB per (b,s); write bf16 [B*S, EMB]
__global__ void k_ln1(const float* X, const float* g, const float* bia, __bf16* Xn) {
  __shared__ float red[256];
  int row = blockIdx.x, tid = threadIdx.x;
  float v = X[(size_t)row * kEMB + tid];
  red[tid] = v; __syncthreads();
  for (int s = 128; s > 0; s >>= 1) { if (tid < s) red[tid] += red[tid + s]; __syncthreads(); }
  float m = red[0] / kEMB; __syncthreads();
  float dv = v - m;
  red[tid] = dv * dv; __syncthreads();
  for (int s = 128; s > 0; s >>= 1) { if (tid < s) red[tid] += red[tid + s]; __syncthreads(); }
  float var = red[0] / kEMB;
  Xn[(size_t)row * kEMB + tid] = f2bf(dv * rsqrtf(var + 1e-6f) * g[tid] + bia[tid]);
}

// qkv[b,o,s] = w_qkv[o,:] . xnorm[b,s,:]  (M=768, N=S, K=256)
__global__ void k_gemm_qkv(const __bf16* Wq, const __bf16* Xn, float* QKV) {
  int lane = threadIdx.x;
  int m0 = blockIdx.x * 16, n0 = blockIdx.y * 16, b = blockIdx.z;
  const __bf16* Bt = Xn + (size_t)b * kS * kEMB;
  v8f acc = {};
  for (int kc = 0; kc < kEMB; kc += 32)
    acc = wmma_bf16(load_a_bf(Wq, kEMB, m0, kc, lane), load_bt_bf(Bt, kEMB, n0, kc, lane), acc);
  int col = lane & 15, rb = ((lane >> 4) & 1) * 8;
#pragma unroll
  for (int r = 0; r < 8; ++r)
    QKV[((size_t)b * 768 + m0 + rb + r) * kS + n0 + col] = acc[r];
}

// depthwise 3x3 SAME on [B,768,7,256]; scatter into q/k f32 [B,H,S,64], v bf16 [B,H,64,S]
__global__ void k_dwconv_qkv(const float* QKV, const float* Wd, float* qf, float* kf, __bf16* vb) {
  int i = blockIdx.x * blockDim.x + threadIdx.x;
  if (i >= kB * 768 * kS) return;
  int s = i % kS, o = (i / kS) % 768, b = i / (768 * kS);
  int vv = s >> 8, wv = s & 255;
  const float* in = QKV + ((size_t)b * 768 + o) * kS;
  float acc = 0.f;
#pragma unroll
  for (int dv = -1; dv <= 1; ++dv)
#pragma unroll
    for (int dw = -1; dw <= 1; ++dw) {
      int v2 = vv + dv, w2 = wv + dw;
      if (v2 >= 0 && v2 < kV && w2 >= 0 && w2 < kW)
        acc += in[v2 * kW + w2] * Wd[o * 9 + (dv + 1) * 3 + (dw + 1)];
    }
  int which = o >> 8, h = (o >> 6) & 3, dd = o & 63;
  if (which == 0)      qf[(((size_t)b * kH + h) * kS + s) * kD + dd] = acc;
  else if (which == 1) kf[(((size_t)b * kH + h) * kS + s) * kD + dd] = acc;
  else                 vb[(((size_t)b * kH + h) * kD + dd) * kS + s] = f2bf(acc);
}

// L2-normalize q,k rows over d=64; write bf16
__global__ void k_qknorm(const float* qf, const float* kf, __bf16* qn, __bf16* kn) {
  int i = blockIdx.x * blockDim.x + threadIdx.x;
  int total = kB * kH * kS;
  if (i >= 2 * total) return;
  const float* src = (i < total) ? qf + (size_t)i * kD : kf + (size_t)(i - total) * kD;
  __bf16*      dst = (i < total) ? qn + (size_t)i * kD : kn + (size_t)(i - total) * kD;
  float ss = 0.f;
  for (int d = 0; d < kD; ++d) { float x = src[d]; ss += x * x; }
  float inv = 1.f / fmaxf(sqrtf(ss), 1e-12f);
  for (int d = 0; d < kD; ++d) dst[d] = f2bf(src[d] * inv);
}

// logits[bh,i,j] = temp[h] * q[i,:].k[j,:]  — wave does 16x64 tile, K=64
__global__ void k_logits(const __bf16* qn, const __bf16* kn, const float* temp, float* att) {
  int lane = threadIdx.x;
  int i0 = blockIdx.x * 16, j0 = blockIdx.y * 64, bh = blockIdx.z;
  const __bf16* Q = qn + (size_t)bh * kS * kD;
  const __bf16* K = kn + (size_t)bh * kS * kD;
  float* A = att + (size_t)bh * kS * kS;
  v8f acc[4] = {};
  for (int kc = 0; kc < kD; kc += 32) {
    v16bf af = load_a_bf(Q, kD, i0, kc, lane);
#pragma unroll
    for (int jt = 0; jt < 4; ++jt)
      acc[jt] = wmma_bf16(af, load_bt_bf(K, kD, j0 + jt * 16, kc, lane), acc[jt]);
  }
  float t = temp[bh & 3];
  int col = lane & 15, rb = ((lane >> 4) & 1) * 8;
#pragma unroll
  for (int jt = 0; jt < 4; ++jt)
#pragma unroll
    for (int r = 0; r < 8; ++r)
      A[(size_t)(i0 + rb + r) * kS + j0 + jt * 16 + col] = acc[jt][r] * t;
}

// per (b,i): softmax + relu^2/gelu branch + cross-head mix; overwrite att in place
__global__ void k_attn_mix(float* att, const float* attca) {
  __shared__ float rows[kH * kS];
  __shared__ float red[256];
  __shared__ float mx[kH], zz[kH], ca[2 * kH * kH];
  int i = blockIdx.x, b = blockIdx.y, tid = threadIdx.x;
  if (tid < 2 * kH * kH) ca[tid] = attca[tid];
  for (int h = 0; h < kH; ++h) {
    const float* r = att + (((size_t)(b * kH + h)) * kS + i) * kS;
    for (int j = tid; j < kS; j += 256) rows[h * kS + j] = r[j];
  }
  __syncthreads();
  for (int h = 0; h < kH; ++h) {
    float m = -3.4e38f;
    for (int j = tid; j < kS; j += 256) m = fmaxf(m, rows[h * kS + j]);
    red[tid] = m; __syncthreads();
    for (int s = 128; s > 0; s >>= 1) { if (tid < s) red[tid] = fmaxf(red[tid], red[tid + s]); __syncthreads(); }
    if (tid == 0) mx[h] = red[0];
    __syncthreads();
    float z = 0.f;
    for (int j = tid; j < kS; j += 256) z += expf(rows[h * kS + j] - mx[h]);
    red[tid] = z; __syncthreads();
    for (int s = 128; s > 0; s >>= 1) { if (tid < s) red[tid] += red[tid + s]; __syncthreads(); }
    if (tid == 0) zz[h] = red[0];
    __syncthreads();
  }
  for (int j = tid; j < kS; j += 256) {
    float l[kH], gg[kH];
#pragma unroll
    for (int h = 0; h < kH; ++h) {
      l[h] = rows[h * kS + j];
      float a1 = fmaxf(l[h], 0.f); a1 *= a1;
      gg[h] = gelu_exact(a1) * a1;
    }
#pragma unroll
    for (int h = 0; h < kH; ++h) {
      float sc = 0.f, sh = 0.f;
#pragma unroll
      for (int h2 = 0; h2 < kH; ++h2) {
        sc += gg[h2] * ca[h * kH + h2];
        sh += gg[h2] * ca[(kH + h) * kH + h2];
      }
      float a0 = expf(l[h] - mx[h]) / zz[h];
      att[(((size_t)(b * kH + h)) * kS + i) * kS + j] = a0 * (1.f + sc) + sh;
    }
  }
}

// out[b,h,i,dd] = attn[i,:].v[:,dd]  (M=S, N=64, K=S); write attO bf16 [B,S,EMB]
__global__ void k_attn_v(const float* att, const __bf16* vb, __bf16* attO) {
  int lane = threadIdx.x;
  int i0 = blockIdx.x * 16, bh = blockIdx.y;
  int b = bh >> 2, h = bh & 3;
  const float* A = att + (size_t)bh * kS * kS;
  const __bf16* Bt = vb + (size_t)bh * kD * kS;   // rows dd, contiguous K=j
  v8f acc[4] = {};
  for (int kc = 0; kc < kS; kc += 32) {
    v16bf af = load_a_f32(A, (size_t)kS, i0, kc, lane);
#pragma unroll
    for (int nt = 0; nt < 4; ++nt)
      acc[nt] = wmma_bf16(af, load_bt_bf(Bt, kS, nt * 16, kc, lane), acc[nt]);
  }
  int col = lane & 15, rb = ((lane >> 4) & 1) * 8;
#pragma unroll
  for (int nt = 0; nt < 4; ++nt)
#pragma unroll
    for (int r = 0; r < 8; ++r)
      attO[((size_t)b * kS + i0 + rb + r) * kEMB + h * kD + nt * 16 + col] = f2bf(acc[nt][r]);
}

// x2[b,o,s] = buffer + w_proj[o,:].attO[b,s,:]  (M=896, N=S, K=256) -> write to d_out
__global__ void k_gemm_proj(const __bf16* Wp, const __bf16* attO, const float* buf, float* out) {
  int lane = threadIdx.x;
  int m0 = blockIdx.x * 16, n0 = blockIdx.y * 16, b = blockIdx.z;
  const __bf16* Bt = attO + (size_t)b * kS * kEMB;
  v8f acc = {};
  for (int kc = 0; kc < kEMB; kc += 32)
    acc = wmma_bf16(load_a_bf(Wp, kEMB, m0, kc, lane), load_bt_bf(Bt, kEMB, n0, kc, lane), acc);
  int col = lane & 15, rb = ((lane >> 4) & 1) * 8;
#pragma unroll
  for (int r = 0; r < 8; ++r) {
    size_t idx = ((size_t)b * kTD + m0 + rb + r) * kS + n0 + col;
    out[idx] = acc[r] + buf[idx];
  }
}

// fused LN2 + GDFN per (b,n) image, fully in LDS; updates d_out in place
__global__ void k_gdfn(float* out, const float* g2, const float* b2,
                       const float* w_pin, const float* w_dw, const float* w_pout) {
  __shared__ float xs[kC * kS];
  __shared__ float es[kC * kS];
  __shared__ float W2[2 * kHF * 9 * kC];   // combined w_dw x w_pin stencil
  __shared__ float ps[kC * kHF];
  int n = blockIdx.x, b = blockIdx.y, tid = threadIdx.x;
  for (int t = tid; t < 2 * kHF * 9 * kC; t += 256) {
    int o2 = t / 36, rem = t % 36, tap = rem / 4, c2 = rem % 4;
    W2[t] = w_dw[o2 * 9 + tap] * w_pin[o2 * 4 + c2];
  }
  for (int t = tid; t < kC * kHF; t += 256) ps[t] = w_pout[t];
  for (int c2 = 0; c2 < kC; ++c2)
    for (int s = tid; s < kS; s += 256)
      xs[c2 * kS + s] = out[((size_t)b * kTD + c2 * kN + n) * kS + s];
  __syncthreads();
  for (int s = tid; s < kS; s += 256) {
    float x0 = xs[s], x1 = xs[kS + s], x2 = xs[2 * kS + s], x3 = xs[3 * kS + s];
    float m = 0.25f * (x0 + x1 + x2 + x3);
    float d0 = x0 - m, d1 = x1 - m, d2 = x2 - m, d3 = x3 - m;
    float inv = rsqrtf(0.25f * (d0 * d0 + d1 * d1 + d2 * d2 + d3 * d3) + 1e-6f);
    es[s]          = d0 * inv * g2[0] + b2[0];
    es[kS + s]     = d1 * inv * g2[1] + b2[1];
    es[2 * kS + s] = d2 * inv * g2[2] + b2[2];
    es[3 * kS + s] = d3 * inv * g2[3] + b2[3];
  }
  __syncthreads();
  for (int s = tid; s < kS; s += 256) {
    int vv = s >> 8, wv = s & 255;
    float nb[9 * kC];
#pragma unroll
    for (int dv = -1; dv <= 1; ++dv)
#pragma unroll
      for (int dw = -1; dw <= 1; ++dw) {
        int tap = (dv + 1) * 3 + (dw + 1);
        int v2 = vv + dv, w2 = wv + dw;
        bool ok = (v2 >= 0 && v2 < kV && w2 >= 0 && w2 < kW);
        int s2 = v2 * kW + w2;
#pragma unroll
        for (int c2 = 0; c2 < kC; ++c2) nb[tap * kC + c2] = ok ? es[c2 * kS + s2] : 0.f;
      }
    float po[kC] = {0.f, 0.f, 0.f, 0.f};
    for (int hf = 0; hf < kHF; ++hf) {
      float s1 = 0.f, s2v = 0.f;
#pragma unroll
      for (int t = 0; t < 36; ++t) {
        s1  += W2[hf * 36 + t] * nb[t];
        s2v += W2[(kHF + hf) * 36 + t] * nb[t];
      }
      float gv = gelu_exact(s1) * s2v;
#pragma unroll
      for (int c2 = 0; c2 < kC; ++c2) po[c2] += ps[c2 * kHF + hf] * gv;
    }
    for (int c2 = 0; c2 < kC; ++c2)
      out[((size_t)b * kTD + c2 * kN + n) * kS + s] += po[c2];
  }
}

} // namespace

extern "C" void kernel_launch(void* const* d_in, const int* in_sizes, int n_in,
                              void* d_out, int out_size, void* d_ws, size_t ws_size,
                              hipStream_t stream) {
  (void)in_sizes; (void)n_in; (void)out_size; (void)ws_size;
  const float* buffer      = (const float*)d_in[0];
  const float* w_in        = (const float*)d_in[1];
  const float* temperature = (const float*)d_in[2];
  const float* ln1_g       = (const float*)d_in[3];
  const float* ln1_b       = (const float*)d_in[4];
  const float* w_qkv       = (const float*)d_in[5];
  const float* w_qkv_dw    = (const float*)d_in[6];
  const float* w_proj      = (const float*)d_in[7];
  const float* w_attca     = (const float*)d_in[8];
  const float* ln2_g       = (const float*)d_in[9];
  const float* ln2_b       = (const float*)d_in[10];
  const float* w_pin       = (const float*)d_in[11];
  const float* w_dw        = (const float*)d_in[12];
  const float* w_pout      = (const float*)d_in[13];
  float* out = (float*)d_out;

  char* w = (char*)d_ws;
  auto alloc = [&](size_t bytes) { void* p = (void*)w; w += (bytes + 255) & ~(size_t)255; return p; };

  __bf16* w_in_bf   = (__bf16*)alloc((size_t)kEMB * kTD * 2);
  __bf16* w_qkv_bf  = (__bf16*)alloc((size_t)768 * kEMB * 2);
  __bf16* w_proj_bf = (__bf16*)alloc((size_t)kTD * kEMB * 2);
  __bf16* tokT      = (__bf16*)alloc((size_t)kRows * kTD * 2);
  float*  xf        = (float*) alloc((size_t)kRows * kEMB * 4);
  __bf16* xnT       = (__bf16*)alloc((size_t)kRows * kEMB * 2);
  float*  qkvf      = (float*) alloc((size_t)kB * 768 * kS * 4);
  float*  qf        = (float*) alloc((size_t)kB * kH * kS * kD * 4);
  float*  kf        = (float*) alloc((size_t)kB * kH * kS * kD * 4);
  __bf16* qn        = (__bf16*)alloc((size_t)kB * kH * kS * kD * 2);
  __bf16* kn        = (__bf16*)alloc((size_t)kB * kH * kS * kD * 2);
  __bf16* vbf       = (__bf16*)alloc((size_t)kB * kH * kS * kD * 2);
  float*  attw      = (float*) alloc((size_t)kB * kH * kS * kS * 4);  // 102.8 MB, lives in L2
  __bf16* attO      = (__bf16*)alloc((size_t)kRows * kEMB * 2);

  int maxw = kEMB * kTD;  // 229376 is the largest weight
  k_cvt3<<<(maxw + 255) / 256, 256, 0, stream>>>(w_in, kEMB * kTD, w_qkv, 768 * kEMB,
                                                 w_proj, kTD * kEMB,
                                                 w_in_bf, w_qkv_bf, w_proj_bf);
  k_tokT<<<(kRows * kTD + 255) / 256, 256, 0, stream>>>(buffer, tokT);
  k_gemm_lin_in<<<dim3(kRows / 16, kEMB / 16), 32, 0, stream>>>(tokT, w_in_bf, xf);
  k_ln1<<<kRows, 256, 0, stream>>>(xf, ln1_g, ln1_b, xnT);
  k_gemm_qkv<<<dim3(768 / 16, kS / 16, kB), 32, 0, stream>>>(w_qkv_bf, xnT, qkvf);
  k_dwconv_qkv<<<(kB * 768 * kS + 255) / 256, 256, 0, stream>>>(qkvf, w_qkv_dw, qf, kf, vbf);
  k_qknorm<<<(2 * kB * kH * kS + 255) / 256, 256, 0, stream>>>(qf, kf, qn, kn);
  k_logits<<<dim3(kS / 16, kS / 64, kB * kH), 32, 0, stream>>>(qn, kn, temperature, attw);
  k_attn_mix<<<dim3(kS, kB), 256, 0, stream>>>(attw, w_attca);
  k_attn_v<<<dim3(kS / 16, kB * kH), 32, 0, stream>>>(attw, vbf, attO);
  k_gemm_proj<<<dim3(kTD / 16, kS / 16, kB), 32, 0, stream>>>(w_proj_bf, attO, buffer, out);
  k_gdfn<<<dim3(kN, kB), 256, 0, stream>>>(out, ln2_g, ln2_b, w_pin, w_dw, w_pout);
}
